// Attention_full_41755672052291
// MI455X (gfx1250) — compile-verified, matrix operand order checked by reading, not run on hardware
//
#include <hip/hip_runtime.h>
#include <hip/hip_bf16.h>
#include <math.h>

#define B_SZ 4
#define CIN  128
#define CO   64
#define NPOS 4096

// LDS leading dims padded for bank-conflict-free 16-lane row gathers
// (dword stride 76 resp. 36 ≡ generators of distinct banks mod 64),
// while keeping each row 16B-aligned so ds_load_b128 merging survives.
#define LD128 152   // for 128-wide (CIN) tiles
#define LD64  72    // for 64-wide (CO / key) tiles

typedef __attribute__((ext_vector_type(16))) __bf16 v16bf;
typedef __attribute__((ext_vector_type(8)))  float  v8f;

union BF16Frag { v16bf v; unsigned int u[8]; };

__device__ __forceinline__ unsigned short f2bf(float f) {
  unsigned int u = __builtin_bit_cast(unsigned int, f);
  u += 0x7FFFu + ((u >> 16) & 1u);           // round-to-nearest-even
  return (unsigned short)(u >> 16);
}
__device__ __forceinline__ unsigned int pack2(float lo, float hi) {
  return (unsigned int)f2bf(lo) | ((unsigned int)f2bf(hi) << 16);
}

// Gather a 16x32 bf16 A-fragment (or, symmetrically, a 32x16 B-fragment)
// from a row-major ushort/bf16 buffer. ISA 7.12.2 16-bit layout:
//   lanes 0-15 : V0..3 = K{0..7}, V4..7 = K{16..23}
//   lanes 16-31: V0..3 = K{8..15}, V4..7 = K{24..31}
__device__ __forceinline__ v16bf load_frag(const unsigned short* __restrict__ base,
                                           int row, int ld, int k0, int lane) {
  const int half = (lane >> 4) & 1;
  BF16Frag f;
#pragma unroll
  for (int p = 0; p < 8; ++p) {
    int k = k0 + 2 * (p & 3) + ((p >= 4) ? 16 : 0) + 8 * half;
    f.u[p] = *(const unsigned int*)(base + row * ld + k);
  }
  return f.v;
}

// ---------------------------------------------------------------------------
// Kernel 1: f/g/h projections.  Per block: one batch, 64 positions.
//   F, G stored as [B][N][CO] bf16 (position-major, contiguous channels)
//   H stored as [B][CO][N] bf16 (channel-major) for the A-side of O = H*P
// ---------------------------------------------------------------------------
__global__ __launch_bounds__(256) void proj_kernel(
    const float* __restrict__ x,
    const float* __restrict__ Wf, const float* __restrict__ bfv,
    const float* __restrict__ Wg, const float* __restrict__ bgv,
    const float* __restrict__ Wh, const float* __restrict__ bhv,
    unsigned short* __restrict__ Fb,
    unsigned short* __restrict__ Gb,
    unsigned short* __restrict__ Hb)
{
  __shared__ unsigned short xs[64][LD128];   // [n][c]
  __shared__ unsigned short wsm[CO][LD128];  // [o][c] (reused per projection)

  const int tid  = threadIdx.x;
  const int lane = tid & 31;
  const int wave = tid >> 5;               // 0..7
  const int b    = blockIdx.x >> 6;
  const int n0   = (blockIdx.x & 63) * 64;
  const int r15  = lane & 15;
  const int half = (lane >> 4) & 1;

  // Stage x tile transposed+converted: x[b][c][n0+n] -> xs[n][c]
  for (int i = tid; i < 64 * CIN; i += 256) {
    int n = i & 63;                        // fastest over n: coalesced global
    int c = i >> 6;
    xs[n][c] = f2bf(x[((size_t)b * CIN + c) * NPOS + n0 + n]);
  }

  const float* Wp[3] = {Wf, Wg, Wh};
  const float* bp[3] = {bfv, bgv, bhv};

  const int band = wave & 3;               // 16-row output-channel band
  const int psub = (wave >> 2) * 2;        // first of two 16-col pos subtiles

#pragma unroll                              // const-folds Wp/bp and the store
  for (int p = 0; p < 3; ++p) {            // target selection below
    __syncthreads();
    for (int i = tid; i < CO * CIN; i += 256) {
      int c = i & 127;
      int o = i >> 7;
      wsm[o][c] = f2bf(Wp[p][o * CIN + c]);
    }
    __syncthreads();

    v16bf afrag[4];
#pragma unroll
    for (int kst = 0; kst < 4; ++kst)
      afrag[kst] = load_frag(&wsm[0][0], band * 16 + r15, LD128, 32 * kst, lane);

    // hoisted bias for this wave's 8 output-channel rows
    float biasr[8];
#pragma unroll
    for (int r = 0; r < 8; ++r)
      biasr[r] = bp[p][band * 16 + r + 8 * half];

#pragma unroll
    for (int s = 0; s < 2; ++s) {
      const int col = (psub + s) * 16 + r15;
      v8f acc = {};
#pragma unroll
      for (int kst = 0; kst < 4; ++kst) {
        v16bf bfrag = load_frag(&xs[0][0], col, LD128, 32 * kst, lane);
        acc = __builtin_amdgcn_wmma_f32_16x16x32_bf16(
            false, afrag[kst], false, bfrag, (short)0, acc, false, false);
      }
      const int n = n0 + col;
#pragma unroll
      for (int r = 0; r < 8; ++r) {
        int o = band * 16 + r + 8 * half;
        unsigned short bv = f2bf(acc[r] + biasr[r]);
        if (p == 0)      Fb[((size_t)b * NPOS + n) * CO + o] = bv;
        else if (p == 1) Gb[((size_t)b * NPOS + n) * CO + o] = bv;
        else             Hb[((size_t)b * CO + o) * NPOS + n] = bv;
      }
    }
  }
}

// ---------------------------------------------------------------------------
// Kernel 2: fused flash attention + final 1x1 conv (Wa, ba, gamma).
// Block = (batch, 128 queries); 8 waves x 16 queries each.
// S^T tiles (rows=keys, cols=queries) => softmax reductions are in-lane +
// one xor-16 shuffle, and P converts purely in-lane to the B-fragment of
// O^T = H^T * P^T.  Same trick feeds O^T into the fused Wa GEMM epilogue.
// Key-tile staging is software-pipelined: tile kt+1's global loads are in
// flight (registers) while tile kt's 16 WMMAs + softmax execute.
// ---------------------------------------------------------------------------
__global__ __launch_bounds__(256) void attn_kernel(
    const unsigned short* __restrict__ Fb,
    const unsigned short* __restrict__ Gb,
    const unsigned short* __restrict__ Hb,
    const float* __restrict__ Wa, const float* __restrict__ bav,
    const float* __restrict__ gammap,
    float* __restrict__ out)
{
  __shared__ unsigned short gsm[64][LD64];   // [key][c]
  __shared__ unsigned short hsm[CO][LD64];   // [co][key]
  __shared__ unsigned short wam[CO][LD64];   // [oc][o]

  const int tid  = threadIdx.x;
  const int lane = tid & 31;
  const int wave = tid >> 5;
  const int b    = blockIdx.x >> 5;
  const int q0   = (blockIdx.x & 31) * 128 + wave * 16;
  const int r15  = lane & 15;
  const int half = (lane >> 4) & 1;

  for (int i = tid; i < CO * CO; i += 256) {
    int o  = i & 63;
    int oc = i >> 6;
    wam[oc][o] = f2bf(Wa[oc * CO + o]);
  }

  // Persistent query (F) B-fragments: K = 64 channels -> 2 fragments
  const unsigned short* fbase = Fb + ((size_t)b * NPOS + q0) * CO;
  v16bf fq[2];
#pragma unroll
  for (int kst = 0; kst < 2; ++kst)
    fq[kst] = load_frag(fbase, r15, CO, 32 * kst, lane);

  float m_run = -3.0e38f;
  float l_run = 0.0f;
  v8f acc[4];
#pragma unroll
  for (int j = 0; j < 4; ++j) acc[j] = v8f{};

  const unsigned short* gBbase = Gb + (size_t)b * NPOS * CO;
  const unsigned short* hBbase = Hb + (size_t)b * CO * NPOS;

  // --- software-pipelined staging: row = one of 64 tile rows, two 16B
  //     chunks per thread (8 uint4 per 128B row, 4 threads per row).
  const int srow = tid >> 2;
  const int sch  = tid & 3;
  uint4 rg0, rg1, rh0, rh1;
  {
    const uint4* gs = (const uint4*)(gBbase + (size_t)srow * CO);
    rg0 = gs[sch]; rg1 = gs[sch + 4];
    const uint4* hs = (const uint4*)(hBbase + (size_t)srow * NPOS);
    rh0 = hs[sch]; rh1 = hs[sch + 4];
  }

  const int NT = NPOS / 64;
  for (int kt = 0; kt < NT; ++kt) {
    // commit the staged tile to LDS
    ((uint4*)&gsm[srow][0])[sch]     = rg0;
    ((uint4*)&gsm[srow][0])[sch + 4] = rg1;
    ((uint4*)&hsm[srow][0])[sch]     = rh0;
    ((uint4*)&hsm[srow][0])[sch + 4] = rh1;
    __syncthreads();

    // issue next tile's global loads; they overlap the compute below
    if (kt + 1 < NT) {
      const int kn = (kt + 1) * 64;
      const uint4* gs = (const uint4*)(gBbase + (size_t)(kn + srow) * CO);
      rg0 = gs[sch]; rg1 = gs[sch + 4];
      const uint4* hs = (const uint4*)(hBbase + (size_t)srow * NPOS + kn);
      rh0 = hs[sch]; rh1 = hs[sch + 4];
    }

    // S^T tiles: rows = 64 keys (4 subtiles), cols = 16 queries, K = 64 ch
    v8f st[4];
#pragma unroll
    for (int sub = 0; sub < 4; ++sub) {
      v8f s = {};
#pragma unroll
      for (int kst = 0; kst < 2; ++kst) {
        v16bf a = load_frag(&gsm[0][0], sub * 16 + r15, LD64, 32 * kst, lane);
        s = __builtin_amdgcn_wmma_f32_16x16x32_bf16(
            false, a, false, fq[kst], (short)0, s, false, false);
      }
      st[sub] = s;
    }

    // Online softmax along keys for this lane's query column
    float mx = st[0][0];
#pragma unroll
    for (int sub = 0; sub < 4; ++sub)
#pragma unroll
      for (int r = 0; r < 8; ++r) mx = fmaxf(mx, st[sub][r]);
    mx = fmaxf(mx, __shfl_xor(mx, 16, 32));
    float m_new = fmaxf(m_run, mx);
    float alpha = __expf(m_run - m_new);
    float lsum = 0.0f;
#pragma unroll
    for (int sub = 0; sub < 4; ++sub)
#pragma unroll
      for (int r = 0; r < 8; ++r) {
        float e = __expf(st[sub][r] - m_new);
        st[sub][r] = e;
        lsum += e;
      }
    lsum += __shfl_xor(lsum, 16, 32);
    l_run = l_run * alpha + lsum;
    m_run = m_new;
#pragma unroll
    for (int j = 0; j < 4; ++j)
#pragma unroll
      for (int r = 0; r < 8; ++r) acc[j][r] *= alpha;

    // In-lane repack of P (C-layout, rows=keys) into two 32x16 B-fragments
    BF16Frag p0, p1;
#pragma unroll
    for (int p = 0; p < 4; ++p) {
      p0.u[p]     = pack2(st[0][2 * p], st[0][2 * p + 1]);
      p0.u[p + 4] = pack2(st[1][2 * p], st[1][2 * p + 1]);
      p1.u[p]     = pack2(st[2][2 * p], st[2][2 * p + 1]);
      p1.u[p + 4] = pack2(st[3][2 * p], st[3][2 * p + 1]);
    }

    // O^T(o, q) += H^T(o, m) * P^T(m, q)
#pragma unroll
    for (int j = 0; j < 4; ++j) {
      v16bf a0 = load_frag(&hsm[0][0], j * 16 + r15, LD64, 0, lane);
      acc[j] = __builtin_amdgcn_wmma_f32_16x16x32_bf16(
          false, a0, false, p0.v, (short)0, acc[j], false, false);
      v16bf a1 = load_frag(&hsm[0][0], j * 16 + r15, LD64, 32, lane);
      acc[j] = __builtin_amdgcn_wmma_f32_16x16x32_bf16(
          false, a1, false, p1.v, (short)0, acc[j], false, false);
    }
    __syncthreads();   // all waves done reading LDS before next commit
  }

  // Softmax normalization
  const float rl = 1.0f / l_run;
#pragma unroll
  for (int j = 0; j < 4; ++j)
#pragma unroll
    for (int r = 0; r < 8; ++r) acc[j][r] *= rl;

  // Fused final conv: out(oc, q) = gamma * (Wa(oc, o) * O^T(o, q) + ba(oc))
  BF16Frag o0, o1;
#pragma unroll
  for (int p = 0; p < 4; ++p) {
    o0.u[p]     = pack2(acc[0][2 * p], acc[0][2 * p + 1]);
    o0.u[p + 4] = pack2(acc[1][2 * p], acc[1][2 * p + 1]);
    o1.u[p]     = pack2(acc[2][2 * p], acc[2][2 * p + 1]);
    o1.u[p + 4] = pack2(acc[3][2 * p], acc[3][2 * p + 1]);
  }
  const float g = gammap[0];
#pragma unroll
  for (int i = 0; i < 4; ++i) {
    v8f ot = {};
    v16bf a0 = load_frag(&wam[0][0], i * 16 + r15, LD64, 0, lane);
    ot = __builtin_amdgcn_wmma_f32_16x16x32_bf16(
        false, a0, false, o0.v, (short)0, ot, false, false);
    v16bf a1 = load_frag(&wam[0][0], i * 16 + r15, LD64, 32, lane);
    ot = __builtin_amdgcn_wmma_f32_16x16x32_bf16(
        false, a1, false, o1.v, (short)0, ot, false, false);
#pragma unroll
    for (int r = 0; r < 8; ++r) {
      int oc = i * 16 + r + 8 * half;
      out[((size_t)b * CO + oc) * NPOS + q0 + r15] = g * (ot[r] + bav[oc]);
    }
  }
}

// ---------------------------------------------------------------------------
extern "C" void kernel_launch(void* const* d_in, const int* in_sizes, int n_in,
                              void* d_out, int out_size, void* d_ws, size_t ws_size,
                              hipStream_t stream) {
  const float* x     = (const float*)d_in[0];
  const float* Wf    = (const float*)d_in[1];
  const float* bf    = (const float*)d_in[2];
  const float* Wg    = (const float*)d_in[3];
  const float* bg    = (const float*)d_in[4];
  const float* Wh    = (const float*)d_in[5];
  const float* bh    = (const float*)d_in[6];
  const float* Wa    = (const float*)d_in[7];
  const float* ba    = (const float*)d_in[8];
  const float* gamma = (const float*)d_in[9];
  float* out = (float*)d_out;

  // Workspace: F, G [B][N][CO] bf16 and H [B][CO][N] bf16  (3 x 2 MB = 6 MB)
  unsigned short* Fb = (unsigned short*)d_ws;
  unsigned short* Gb = Fb + (size_t)B_SZ * NPOS * CO;
  unsigned short* Hb = Gb + (size_t)B_SZ * NPOS * CO;

  proj_kernel<<<B_SZ * (NPOS / 64), 256, 0, stream>>>(
      x, Wf, bf, Wg, bg, Wh, bh, Fb, Gb, Hb);
  attn_kernel<<<B_SZ * (NPOS / 128), 256, 0, stream>>>(
      Fb, Gb, Hb, Wa, ba, gamma, out);
}
